// MultiHeadAttention_22385369547030
// MI455X (gfx1250) — compile-verified, matrix operand order checked by reading, not run on hardware
//
#include <hip/hip_runtime.h>

// ---------- types ----------
typedef __bf16 bf16_t;
typedef __attribute__((ext_vector_type(16))) __bf16 v16bf;
typedef __attribute__((ext_vector_type(8)))  __bf16 v8bf;
typedef __attribute__((ext_vector_type(8)))  float  v8f;

union Frag { v16bf v; v8bf h[2]; };

#if defined(__HIP_DEVICE_COMPILE__) && defined(__has_builtin)
#if __has_builtin(__builtin_amdgcn_global_load_async_to_lds_b128)
#define HAVE_ASYNC_LDS 1
#endif
#endif

#ifdef HAVE_ASYNC_LDS
// Builtin signature (from clang diagnostic): param0 = v4i addrspace(1)*,
// param1 = v4i addrspace(3)*, then two imm ints (offset, cpol).
typedef int v4i __attribute__((vector_size(4 * sizeof(int))));
typedef __attribute__((address_space(1))) v4i gbl_v4i;
typedef __attribute__((address_space(3))) v4i lds_v4i;
__device__ __forceinline__ gbl_v4i* to_gbl(const void* p) {
    return (gbl_v4i*)(unsigned long long)(uintptr_t)p;
}
__device__ __forceinline__ lds_v4i* to_lds(void* p) {
    return (lds_v4i*)(unsigned int)(uintptr_t)p;  // flat LDS addr low bits = LDS offset
}
#endif

__device__ __forceinline__ v8f zero8() {
    v8f z;
#pragma unroll
    for (int i = 0; i < 8; ++i) z[i] = 0.0f;
    return z;
}

__device__ __forceinline__ v8bf pack8(const float4& a, const float4& b) {
    v8bf r;
    r[0] = (__bf16)a.x; r[1] = (__bf16)a.y; r[2] = (__bf16)a.z; r[3] = (__bf16)a.w;
    r[4] = (__bf16)b.x; r[5] = (__bf16)b.y; r[6] = (__bf16)b.z; r[7] = (__bf16)b.w;
    return r;
}

// =====================================================================
// Tiled bf16 WMMA GEMM:  Y[M,N] = X[M,K] @ W[N,K]^T + bias[N]
// Block tile 128x64, K-step 32. 256 threads = 8 waves, wave tile 32x32.
// EPI 0: f32 out, row-major [M,N]
// EPI 1: bf16 out, [B,H,S,Dh]   (n = h*64+d, m = b*S+s)
// EPI 2: bf16 out, [B,H,Dh,S]   (V transposed for attention B-operand)
// =====================================================================
template <int EPI, typename IN_T>
__global__ __launch_bounds__(256) void gemm_wmma(
    const IN_T* __restrict__ X, const float* __restrict__ W,
    const float* __restrict__ bias, void* __restrict__ out,
    int M, int N, int K, int Sdim, int Hdim)
{
    constexpr int BM = 128, BN = 64, BK = 32, LD = 40; // LD: padded row (elems)
    __shared__ __bf16 As[BM][LD];
    __shared__ __bf16 Bs[BN][LD];

    const int tid  = threadIdx.x;
    const int lane = tid & 31;
    const int wid  = tid >> 5;
    const int lm   = lane & 15;     // col-within-tile / row-of-A-frag
    const int lg   = lane >> 4;     // half-wave group
    const int wm   = (wid & 3) * 32;
    const int wn   = (wid >> 2) * 32;
    const int bm   = blockIdx.x * BM;
    const int bn   = blockIdx.y * BN;

    v8f acc[2][2];
#pragma unroll
    for (int i = 0; i < 2; ++i)
#pragma unroll
        for (int j = 0; j < 2; ++j) acc[i][j] = zero8();

    // staging coordinates
    const int arow = tid >> 1, acol = (tid & 1) * 16;   // X: 16 elems/thread
    const int brow = tid >> 2, bcol = (tid & 3) * 8;    // W: 8 elems/thread

    for (int k0 = 0; k0 < K; k0 += BK) {
        // ---- stage X tile (convert to bf16 / async copy if already bf16) ----
        {
            const IN_T* src = X + (size_t)(bm + arow) * K + k0 + acol;
            if (k0 + BK < K)
                __builtin_prefetch(src + BK, 0, 1);     // global_prefetch_b8
            if constexpr (sizeof(IN_T) == 4) {          // fp32 input: convert
                const float4* s4 = reinterpret_cast<const float4*>(src);
                float4 f0 = s4[0], f1 = s4[1], f2 = s4[2], f3 = s4[3];
                *(v8bf*)&As[arow][acol]     = pack8(f0, f1);
                *(v8bf*)&As[arow][acol + 8] = pack8(f2, f3);
            } else {                                    // bf16 input: pure copy
#ifdef HAVE_ASYNC_LDS
                __builtin_amdgcn_global_load_async_to_lds_b128(
                    to_gbl(src),     to_lds(&As[arow][acol]),     0, 0);
                __builtin_amdgcn_global_load_async_to_lds_b128(
                    to_gbl(src + 8), to_lds(&As[arow][acol + 8]), 0, 0);
#else
                const v8bf* s8 = reinterpret_cast<const v8bf*>(src);
                *(v8bf*)&As[arow][acol]     = s8[0];
                *(v8bf*)&As[arow][acol + 8] = s8[1];
#endif
            }
        }
        // ---- stage W tile (convert to bf16) ----
        {
            const float4* s4 = reinterpret_cast<const float4*>(
                W + (size_t)(bn + brow) * K + k0 + bcol);
            float4 f0 = s4[0], f1 = s4[1];
            *(v8bf*)&Bs[brow][bcol] = pack8(f0, f1);
        }
#ifdef HAVE_ASYNC_LDS
        if constexpr (sizeof(IN_T) == 2)
            asm volatile("s_wait_asynccnt 0" ::: "memory");
#endif
        __syncthreads();

        // ---- fragments (documented 16-bit A layout: k = idx+(idx>=8?8:0)+8g) ----
        Frag a[2], b[2];
#pragma unroll
        for (int t = 0; t < 2; ++t) {
            const __bf16* ap = &As[wm + t * 16 + lm][0];
            a[t].h[0] = *(const v8bf*)(ap + 8 * lg);
            a[t].h[1] = *(const v8bf*)(ap + 16 + 8 * lg);
            const __bf16* bp = &Bs[wn + t * 16 + lm][0];
            b[t].h[0] = *(const v8bf*)(bp + 8 * lg);
            b[t].h[1] = *(const v8bf*)(bp + 16 + 8 * lg);
        }
#pragma unroll
        for (int tm = 0; tm < 2; ++tm)
#pragma unroll
            for (int tn = 0; tn < 2; ++tn)
                acc[tm][tn] = __builtin_amdgcn_wmma_f32_16x16x32_bf16(
                    false, a[tm].v, false, b[tn].v, (short)0, acc[tm][tn],
                    false, false);
        __syncthreads();
    }

    // ---- epilogue: C layout row m = r + 8*(lane/16), col n = lane%16 ----
#pragma unroll
    for (int tm = 0; tm < 2; ++tm)
#pragma unroll
        for (int tn = 0; tn < 2; ++tn) {
            const int ng = bn + wn + tn * 16 + lm;
            const float bv = bias[ng];
#pragma unroll
            for (int r = 0; r < 8; ++r) {
                const int mg = bm + wm + tm * 16 + r + 8 * lg;
                const float val = acc[tm][tn][r] + bv;
                if constexpr (EPI == 0) {
                    ((float*)out)[(size_t)mg * N + ng] = val;
                } else if constexpr (EPI == 1) {
                    const int h = ng >> 6, d = ng & 63;
                    const int bb = mg / Sdim, ss = mg % Sdim;
                    ((bf16_t*)out)[(((size_t)bb * Hdim + h) * Sdim + ss) * 64 + d] =
                        (__bf16)val;
                } else {
                    const int h = ng >> 6, d = ng & 63;
                    const int bb = mg / Sdim, ss = mg % Sdim;
                    ((bf16_t*)out)[(((size_t)bb * Hdim + h) * 64 + d) * Sdim + ss] =
                        (__bf16)val;
                }
            }
        }
}

// =====================================================================
// Flash-attention: one (b,h) x 64 Q-rows per block, 4 waves x 16 rows.
// Q,K: bf16 [B,H,S,64]; Vt: bf16 [B,H,64,S]; out: bf16 [B,S,H*64]
// =====================================================================
__global__ __launch_bounds__(128) void attn_wmma(
    const bf16_t* __restrict__ Q, const bf16_t* __restrict__ Kd,
    const bf16_t* __restrict__ Vt, bf16_t* __restrict__ Oo,
    int Bn, int Hn, int Sn)
{
    constexpr int DH = 64, CT = 64, LD = 72;
    __shared__ __bf16 Ks[CT][LD];        // [t][d]
    __shared__ __bf16 Vs[DH][LD];        // [d][t]  (V pre-transposed in HBM)
    __shared__ __bf16 Ps[4][16][LD];     // per-wave P tile

    const int tid  = threadIdx.x;
    const int lane = tid & 31;
    const int wid  = tid >> 5;
    const int lm   = lane & 15;
    const int lg   = lane >> 4;
    const int bh   = blockIdx.y;
    const int b    = bh / Hn, h = bh % Hn;
    const int q0   = blockIdx.x * 64;

    // Q fragments for this wave's 16 rows (resident across the whole T loop)
    const bf16_t* Qb = Q + ((size_t)bh * Sn + q0 + wid * 16) * DH;
    Frag qf[2];
#pragma unroll
    for (int ks = 0; ks < 2; ++ks) {
        const bf16_t* qp = Qb + (size_t)lm * DH + ks * 32;
        qf[ks].h[0] = *(const v8bf*)(qp + 8 * lg);
        qf[ks].h[1] = *(const v8bf*)(qp + 16 + 8 * lg);
    }

    v8f oacc[4];
    float mrun[8], lrun[8];
#pragma unroll
    for (int d = 0; d < 4; ++d) oacc[d] = zero8();
#pragma unroll
    for (int r = 0; r < 8; ++r) { mrun[r] = -1e30f; lrun[r] = 0.0f; }

    const bf16_t* Kb = Kd + (size_t)bh * Sn * DH;
    const bf16_t* Vb = Vt + (size_t)bh * DH * Sn;
    const int srow = tid >> 1, scol = (tid & 1) * 32;   // staging coords
    const float scale = 0.125f;                         // 1/sqrt(64)

    for (int t0 = 0; t0 < Sn; t0 += CT) {
        __syncthreads();   // previous chunk's LDS reads done
        // ---- stage K chunk [64 x 64] and V^T chunk [64 x 64] ----
        {
            const bf16_t* ksrc = Kb + (size_t)(t0 + srow) * DH + scol;
            const bf16_t* vsrc = Vb + (size_t)srow * Sn + t0 + scol;
#ifdef HAVE_ASYNC_LDS
#pragma unroll
            for (int i = 0; i < 4; ++i) {
                __builtin_amdgcn_global_load_async_to_lds_b128(
                    to_gbl(ksrc + i * 8), to_lds(&Ks[srow][scol + i * 8]), 0, 0);
                __builtin_amdgcn_global_load_async_to_lds_b128(
                    to_gbl(vsrc + i * 8), to_lds(&Vs[srow][scol + i * 8]), 0, 0);
            }
            asm volatile("s_wait_asynccnt 0" ::: "memory");
#else
            const uint4* k4 = (const uint4*)ksrc;
            const uint4* v4 = (const uint4*)vsrc;
#pragma unroll
            for (int i = 0; i < 4; ++i) {
                *(uint4*)&Ks[srow][scol + i * 8] = k4[i];
                *(uint4*)&Vs[srow][scol + i * 8] = v4[i];
            }
#endif
        }
        __syncthreads();

        // ---- scores: S = Q @ K^T  (4 tiles of 16x16 per wave) ----
        v8f sc[4];
#pragma unroll
        for (int t = 0; t < 4; ++t) sc[t] = zero8();
#pragma unroll
        for (int ks = 0; ks < 2; ++ks) {
            Frag kb[4];
#pragma unroll
            for (int tn = 0; tn < 4; ++tn) {
                const __bf16* bp = &Ks[tn * 16 + lm][ks * 32];
                kb[tn].h[0] = *(const v8bf*)(bp + 8 * lg);
                kb[tn].h[1] = *(const v8bf*)(bp + 16 + 8 * lg);
            }
#pragma unroll
            for (int tn = 0; tn < 4; ++tn)
                sc[tn] = __builtin_amdgcn_wmma_f32_16x16x32_bf16(
                    false, qf[ks].v, false, kb[tn].v, (short)0, sc[tn],
                    false, false);
        }

        // ---- online softmax (row = r + 8*lg, reduce over 16-lane groups) ----
#pragma unroll
        for (int r = 0; r < 8; ++r) {
            float cm = -1e30f;
#pragma unroll
            for (int tn = 0; tn < 4; ++tn) cm = fmaxf(cm, sc[tn][r]);
            cm *= scale;
#pragma unroll
            for (int off = 1; off < 16; off <<= 1)
                cm = fmaxf(cm, __shfl_xor(cm, off, 32));
            const float mnew  = fmaxf(mrun[r], cm);
            const float alpha = __expf(mrun[r] - mnew);
            float ls = 0.0f;
#pragma unroll
            for (int tn = 0; tn < 4; ++tn) {
                const float p = __expf(sc[tn][r] * scale - mnew);
                ls += p;
                Ps[wid][r + 8 * lg][tn * 16 + lm] = (__bf16)p;
            }
#pragma unroll
            for (int off = 1; off < 16; off <<= 1)
                ls += __shfl_xor(ls, off, 32);
            lrun[r] = lrun[r] * alpha + ls;
            mrun[r] = mnew;
#pragma unroll
            for (int dt = 0; dt < 4; ++dt) oacc[dt][r] *= alpha;
        }

        // ---- O += P @ V  (P reloaded from own wave's LDS in A layout) ----
#pragma unroll
        for (int ks = 0; ks < 2; ++ks) {
            Frag pf;
            const __bf16* pp = &Ps[wid][lm][ks * 32];
            pf.h[0] = *(const v8bf*)(pp + 8 * lg);
            pf.h[1] = *(const v8bf*)(pp + 16 + 8 * lg);
#pragma unroll
            for (int dt = 0; dt < 4; ++dt) {
                Frag vf;
                const __bf16* vp = &Vs[dt * 16 + lm][ks * 32];
                vf.h[0] = *(const v8bf*)(vp + 8 * lg);
                vf.h[1] = *(const v8bf*)(vp + 16 + 8 * lg);
                oacc[dt] = __builtin_amdgcn_wmma_f32_16x16x32_bf16(
                    false, pf.v, false, vf.v, (short)0, oacc[dt], false, false);
            }
        }
    }

    // ---- normalize + scatter to [B,S,H*64] (bf16, feeds output GEMM) ----
#pragma unroll
    for (int dt = 0; dt < 4; ++dt)
#pragma unroll
        for (int r = 0; r < 8; ++r) {
            const float val = oacc[dt][r] / lrun[r];
            const size_t row = (size_t)b * Sn + q0 + wid * 16 + r + 8 * lg;
            Oo[row * (size_t)(Hn * 64) + h * 64 + dt * 16 + lm] = (__bf16)val;
        }
}

// =====================================================================
extern "C" void kernel_launch(void* const* d_in, const int* in_sizes, int n_in,
                              void* d_out, int out_size, void* d_ws, size_t ws_size,
                              hipStream_t stream) {
    (void)in_sizes; (void)n_in; (void)out_size; (void)ws_size;
    constexpr int B = 4, S = 2048, E = 1024, H = 16, DH = 64;
    constexpr int M = B * S, N = H * DH;

    const float* x  = (const float*)d_in[0];
    const float* Wq = (const float*)d_in[1];
    const float* bq = (const float*)d_in[2];
    const float* Wk = (const float*)d_in[3];
    const float* bk = (const float*)d_in[4];
    const float* Wv = (const float*)d_in[5];
    const float* bv = (const float*)d_in[6];
    const float* Wo = (const float*)d_in[7];
    const float* bo = (const float*)d_in[8];

    const size_t QN = (size_t)B * H * S * DH;   // 8,388,608 elems
    bf16_t* Qw = (bf16_t*)d_ws;                 // [B,H,S,DH]
    bf16_t* Kw = Qw + QN;                       // [B,H,S,DH]
    bf16_t* Vw = Kw + QN;                       // [B,H,DH,S] (transposed)
    bf16_t* Aw = Vw + QN;                       // [B,S,H*DH]

    dim3 gG(M / 128, N / 64);
    gemm_wmma<1, float><<<gG, 256, 0, stream>>>(x, Wq, bq, Qw, M, N, E, S, H);
    gemm_wmma<1, float><<<gG, 256, 0, stream>>>(x, Wk, bk, Kw, M, N, E, S, H);
    gemm_wmma<2, float><<<gG, 256, 0, stream>>>(x, Wv, bv, Vw, M, N, E, S, H);

    attn_wmma<<<dim3(S / 64, B * H), 128, 0, stream>>>(Qw, Kw, Vw, Aw, B, H, S);

    gemm_wmma<0, bf16_t><<<dim3(M / 128, E / 64), 256, 0, stream>>>(
        Aw, Wo, bo, d_out, M, E, N, S, H);
}
